// CandidateScorer_63299228009096
// MI455X (gfx1250) — matrix operand order repaired and checked by reading, not compiled, not run on hardware
//
#include <hip/hip_runtime.h>

// ---------------------------------------------------------------------------
// Types for CDNA5 WMMA (wave32): v_wmma_f32_16x16x32_bf16
// ---------------------------------------------------------------------------
typedef __attribute__((ext_vector_type(16))) __bf16 v16bf;
typedef __attribute__((ext_vector_type(8)))  __bf16 v8bf;
typedef __attribute__((ext_vector_type(8)))  float  v8f;

union Frag {
    v16bf v;
    v8bf  h[2];
};

#define MT   64      // candidates per block (GEMM M-tile, 4 WMMA M-tiles)
#define BLK  512     // 16 waves; each wave owns 64 output columns (4 N-tiles)
#define IN_K 832     // input dim = 256 + 256 + 256 + 64
#define HID  1024
#define LDA  840     // LDS stride for X tile  (16B-aligned rows)
#define LDH  1032    // LDS stride for H tiles (16B-aligned rows)

__device__ __forceinline__ unsigned short f2bf_bits(float x) {
    __bf16 b = (__bf16)x;
    return __builtin_bit_cast(unsigned short, b);
}

// ---------------------------------------------------------------------------
// Kernel 1: global mean pool over 19x19 per channel
// ---------------------------------------------------------------------------
__global__ void k_pool(const float* __restrict__ pf, float* __restrict__ gp) {
    int ch = threadIdx.x;
    if (ch < 256) {
        const float* p = pf + ch * 361;
        float s = 0.f;
        #pragma unroll 19
        for (int i = 0; i < 361; ++i) s += p[i];
        gp[ch] = s * (1.0f / 361.0f);
    }
}

// ---------------------------------------------------------------------------
// Kernel 2: weight transpose + f32 -> bf16.  in: (K,N) row-major, out: (N,K)
// N-major so each WMMA B-fragment (two runs of 8 K of one column) is two
// contiguous 16B global loads per lane.
// ---------------------------------------------------------------------------
__global__ void k_transpose_to_bf16(const float* __restrict__ in,
                                    unsigned short* __restrict__ out,
                                    int K, int N) {
    int total = K * N;
    for (int idx = blockIdx.x * blockDim.x + threadIdx.x; idx < total;
         idx += gridDim.x * blockDim.x) {
        int n = idx / K;
        int k = idx - n * K;
        out[idx] = f2bf_bits(in[(size_t)k * N + n]);
    }
}

// ---------------------------------------------------------------------------
// One fused GEMM layer: Os(64 x 1024, bf16) = relu(As(64 x K, bf16) @ Wt^T + b)
// Wt is (1024 x K) bf16, N-major.  Wave owns columns [wave*64, wave*64+64),
// processed as 2 pairs of N-tiles so each A-fragment LDS load feeds 2 WMMAs.
//
// A-fragment (ISA 7.12.2, 16-bit A 16x32): lane&15 = M row,
//   lane<16 -> K {0..7,16..23}, lane>=16 -> K {8..15,24..31}.
// B-fragment: lane&15 = N col, lane<16 -> K {0..7,16..23}, lane>=16 -> K {8..15,24..31}.
// C layout: VGPR r -> M=r (lanes 0-15) / M=8+r (lanes 16-31), N = lane&15.
// ---------------------------------------------------------------------------
__device__ __forceinline__ void gemm_layer(const __bf16* __restrict__ As, int lda, int K,
                                           const __bf16* __restrict__ Wt, int kd,
                                           const float* __restrict__ bias,
                                           __bf16* __restrict__ Os, int ldo,
                                           int lane, int wave) {
    const int half = lane >> 4;   // 0 or 1
    const int l15  = lane & 15;
    const int mlo  = half * 8;    // C-fragment row base inside a 16-row tile

    const __bf16* aptr0 = As + l15 * lda + half * 8;   // M-tile 0 (rows  0..15)
    const __bf16* aptr1 = aptr0 + 16 * lda;            // M-tile 1 (rows 16..31)
    const __bf16* aptr2 = aptr0 + 32 * lda;            // M-tile 2 (rows 32..47)
    const __bf16* aptr3 = aptr0 + 48 * lda;            // M-tile 3 (rows 48..63)

    #pragma unroll
    for (int np = 0; np < 2; ++np) {                   // pair of N-tiles
        const int n0 = wave * 64 + np * 32 + l15;      // this lane's columns
        const int n1 = n0 + 16;
        const __bf16* bptr0 = Wt + (size_t)n0 * kd + half * 8;
        const __bf16* bptr1 = Wt + (size_t)n1 * kd + half * 8;

        v8f a0n0 = {}, a1n0 = {}, a2n0 = {}, a3n0 = {};
        v8f a0n1 = {}, a1n1 = {}, a2n1 = {}, a3n1 = {};

        for (int k0 = 0; k0 < K; k0 += 32) {
            Frag a0, a1, a2, a3, b0, b1;
            a0.h[0] = *(const v8bf*)(aptr0 + k0);
            a0.h[1] = *(const v8bf*)(aptr0 + k0 + 16);
            a1.h[0] = *(const v8bf*)(aptr1 + k0);
            a1.h[1] = *(const v8bf*)(aptr1 + k0 + 16);
            a2.h[0] = *(const v8bf*)(aptr2 + k0);
            a2.h[1] = *(const v8bf*)(aptr2 + k0 + 16);
            a3.h[0] = *(const v8bf*)(aptr3 + k0);
            a3.h[1] = *(const v8bf*)(aptr3 + k0 + 16);
            b0.h[0] = *(const v8bf*)(bptr0 + k0);
            b0.h[1] = *(const v8bf*)(bptr0 + k0 + 16);
            b1.h[0] = *(const v8bf*)(bptr1 + k0);
            b1.h[1] = *(const v8bf*)(bptr1 + k0 + 16);
            __builtin_prefetch(bptr0 + k0 + 256, 0, 1);   // global_prefetch_b8
            __builtin_prefetch(bptr1 + k0 + 256, 0, 1);

            a0n0 = __builtin_amdgcn_wmma_f32_16x16x32_bf16(false, a0.v, false, b0.v, (short)0, a0n0, false, false);
            a1n0 = __builtin_amdgcn_wmma_f32_16x16x32_bf16(false, a1.v, false, b0.v, (short)0, a1n0, false, false);
            a2n0 = __builtin_amdgcn_wmma_f32_16x16x32_bf16(false, a2.v, false, b0.v, (short)0, a2n0, false, false);
            a3n0 = __builtin_amdgcn_wmma_f32_16x16x32_bf16(false, a3.v, false, b0.v, (short)0, a3n0, false, false);
            a0n1 = __builtin_amdgcn_wmma_f32_16x16x32_bf16(false, a0.v, false, b1.v, (short)0, a0n1, false, false);
            a1n1 = __builtin_amdgcn_wmma_f32_16x16x32_bf16(false, a1.v, false, b1.v, (short)0, a1n1, false, false);
            a2n1 = __builtin_amdgcn_wmma_f32_16x16x32_bf16(false, a2.v, false, b1.v, (short)0, a2n1, false, false);
            a3n1 = __builtin_amdgcn_wmma_f32_16x16x32_bf16(false, a3.v, false, b1.v, (short)0, a3n1, false, false);
        }

        const float bv0 = bias[n0];
        const float bv1 = bias[n1];
        #pragma unroll
        for (int r = 0; r < 8; ++r) {
            float v;
            v = a0n0[r] + bv0; v = v > 0.f ? v : 0.f; Os[(     mlo + r) * ldo + n0] = (__bf16)v;
            v = a1n0[r] + bv0; v = v > 0.f ? v : 0.f; Os[(16 + mlo + r) * ldo + n0] = (__bf16)v;
            v = a2n0[r] + bv0; v = v > 0.f ? v : 0.f; Os[(32 + mlo + r) * ldo + n0] = (__bf16)v;
            v = a3n0[r] + bv0; v = v > 0.f ? v : 0.f; Os[(48 + mlo + r) * ldo + n0] = (__bf16)v;
            v = a0n1[r] + bv1; v = v > 0.f ? v : 0.f; Os[(     mlo + r) * ldo + n1] = (__bf16)v;
            v = a1n1[r] + bv1; v = v > 0.f ? v : 0.f; Os[(16 + mlo + r) * ldo + n1] = (__bf16)v;
            v = a2n1[r] + bv1; v = v > 0.f ? v : 0.f; Os[(32 + mlo + r) * ldo + n1] = (__bf16)v;
            v = a3n1[r] + bv1; v = v > 0.f ? v : 0.f; Os[(48 + mlo + r) * ldo + n1] = (__bf16)v;
        }
    }
}

// ---------------------------------------------------------------------------
// Kernel 3: fused gather + 3-layer MLP.  One block = 64 candidates, 16 waves.
// ---------------------------------------------------------------------------
__global__ void __launch_bounds__(BLK)
k_fused(const float* __restrict__ pf,
        const float* __restrict__ cand,
        const float* __restrict__ b1,
        const float* __restrict__ b2,
        const float* __restrict__ b3,
        const float* __restrict__ gp,
        const unsigned short* __restrict__ w1t_u,
        const unsigned short* __restrict__ w2t_u,
        const unsigned short* __restrict__ w3_u,
        float* __restrict__ out) {
    extern __shared__ char smem[];
    __bf16* bufA = (__bf16*)smem;          // X tile (stride LDA), later H2 (stride LDH)
    __bf16* bufB = bufA + MT * LDH;        // H1 (stride LDH)
    float*  red  = (float*)(bufB + MT * LDH);
    __shared__ int s_go[MT];
    __shared__ int s_to[MT];

    const __bf16* W1t = (const __bf16*)w1t_u;
    const __bf16* W2t = (const __bf16*)w2t_u;
    const __bf16* W3b = (const __bf16*)w3_u;

    const int tid  = threadIdx.x;
    const int lane = tid & 31;
    const int wave = tid >> 5;
    const int c0   = blockIdx.x * MT;

    // Decode board coordinates once per candidate row
    if (tid < MT) {
        const float* cf = cand + (size_t)(c0 + tid) * 64;
        int gr = (int)(cf[5] * 18.f); gr = gr < 0 ? 0 : (gr > 18 ? 18 : gr);
        int gc = (int)(cf[6] * 18.f); gc = gc < 0 ? 0 : (gc > 18 ? 18 : gc);
        int tr = (int)(cf[7] * 18.f); tr = tr < 0 ? 0 : (tr > 18 ? 18 : tr);
        int tc = (int)(cf[8] * 18.f); tc = tc < 0 ? 0 : (tc > 18 ? 18 : tc);
        s_go[tid] = gr * 19 + gc;
        s_to[tid] = tr * 19 + tc;
        red[tid]  = 0.f;
    }
    __syncthreads();

    // Build combined input tile X (64 x 832) in LDS as bf16
    for (int idx = tid; idx < MT * IN_K; idx += BLK) {
        int r   = idx / IN_K;
        int col = idx - r * IN_K;
        float v;
        if (col < 256)      v = pf[col * 361 + s_go[r]];
        else if (col < 512) v = pf[(col - 256) * 361 + s_to[r]];
        else if (col < 768) v = gp[col - 512];
        else                v = cand[(size_t)(c0 + r) * 64 + (col - 768)];
        bufA[r * LDA + col] = (__bf16)v;
    }
    __syncthreads();

    // Layer 1: H1 = relu(X @ W1 + b1)      (K = 832 = 26 WMMA steps)
    gemm_layer(bufA, LDA, IN_K, W1t, IN_K, b1, bufB, LDH, lane, wave);
    __syncthreads();

    // Layer 2: H2 = relu(H1 @ W2 + b2)     (K = 1024 = 32 WMMA steps), H2 reuses bufA
    gemm_layer(bufB, LDH, HID, W2t, HID, b2, bufA, LDH, lane, wave);
    __syncthreads();

    // Layer 3: scores = H2 @ W3 + b3  (N=1 -> VALU dot + LDS f32 atomic reduce)
    {
        const int row  = tid & 63;
        const int part = tid >> 6;               // 8 partials of 128 each
        const __bf16* h2 = bufA + row * LDH + part * 128;
        const __bf16* w3 = W3b + part * 128;
        float s = 0.f;
        #pragma unroll 8
        for (int k = 0; k < 128; ++k) s += (float)h2[k] * (float)w3[k];
        atomicAdd(&red[row], s);                 // ds_add_f32
    }
    __syncthreads();
    if (tid < MT) out[c0 + tid] = red[tid] + b3[0];
}

// ---------------------------------------------------------------------------
// Host launcher
// ---------------------------------------------------------------------------
extern "C" void kernel_launch(void* const* d_in, const int* in_sizes, int n_in,
                              void* d_out, int out_size, void* d_ws, size_t ws_size,
                              hipStream_t stream) {
    const float* pf   = (const float*)d_in[0];   // (256, 19, 19)
    const float* cand = (const float*)d_in[1];   // (N, 64)
    const float* W1   = (const float*)d_in[2];   // (832, 1024)
    const float* b1   = (const float*)d_in[3];   // (1024,)
    const float* W2   = (const float*)d_in[4];   // (1024, 1024)
    const float* b2   = (const float*)d_in[5];   // (1024,)
    const float* W3   = (const float*)d_in[6];   // (1024, 1)
    const float* b3   = (const float*)d_in[7];   // (1,)
    float* out = (float*)d_out;

    const int N = in_sizes[1] / 64;              // 65536

    // Workspace layout (bf16 weights stored N-major, plus pooled features)
    char* ws = (char*)d_ws;
    float*          gp  = (float*)ws;                                      // 1 KB
    unsigned short* w1t = (unsigned short*)(ws + 1024);                    // 1024*832*2
    unsigned short* w2t = (unsigned short*)(ws + 1024 + 1024 * 832 * 2);   // 1024*1024*2
    unsigned short* w3b = (unsigned short*)(ws + 1024 + 1024 * 832 * 2
                                               + 1024 * 1024 * 2);         // 1024*2

    k_pool<<<1, 256, 0, stream>>>(pf, gp);
    k_transpose_to_bf16<<<(832 * 1024 + 255) / 256, 256, 0, stream>>>(W1, w1t, 832, 1024);
    k_transpose_to_bf16<<<(1024 * 1024 + 255) / 256, 256, 0, stream>>>(W2, w2t, 1024, 1024);
    k_transpose_to_bf16<<<(1024 + 255) / 256, 256, 0, stream>>>(W3, w3b, 1024, 1);

    const size_t smem = (size_t)(2 * MT * LDH) * sizeof(unsigned short)
                      + (size_t)MT * sizeof(float);      // ~264 KB of 320 KB WGP LDS
    k_fused<<<N / MT, BLK, smem, stream>>>(pf, cand, b1, b2, b3, gp,
                                           w1t, w2t, w3b, out);
}
